// BinaryConnectNet_52158082843185
// MI455X (gfx1250) — matrix operand order repaired and checked by reading, not compiled, not run on hardware
//
#include <hip/hip_runtime.h>

typedef __attribute__((ext_vector_type(16))) _Float16 v16h;
typedef __attribute__((ext_vector_type(8)))  _Float16 v8h;
typedef __attribute__((ext_vector_type(8)))  float    v8f;

__device__ __forceinline__ float fsign(float x) {
    return (x > 0.f) ? 1.f : ((x < 0.f) ? -1.f : 0.f);
}

// ---------------------------------------------------------------------------
// K0: sign(w2_pw) -> f16  (256*128 = 32768 elems)
// ---------------------------------------------------------------------------
__global__ __launch_bounds__(256) void k0_signw2(const float* __restrict__ w,
                                                 _Float16* __restrict__ ws) {
    int i = blockIdx.x * 256 + threadIdx.x;
    ws[i] = (_Float16)fsign(w[i]);
}

// ---------------------------------------------------------------------------
// K5: fc1_w fp32 -> f16  (1024*16384 elems)
// ---------------------------------------------------------------------------
__global__ __launch_bounds__(256) void k5_cvt(const float* __restrict__ w,
                                              _Float16* __restrict__ wh, int n) {
    int i = blockIdx.x * 256 + threadIdx.x;
    if (i < n) wh[i] = (_Float16)w[i];
}

// ---------------------------------------------------------------------------
// K1: conv1_dw(3x3, g=3) + conv1_pw(1x1, 3->128) + sign + maxpool2
//     one block per image; t[3][32][32] staged in LDS
//     out a1: [N][128][16][16] in {-1,0,+1} as f16
// ---------------------------------------------------------------------------
__global__ __launch_bounds__(256) void k1_conv1(
    const float* __restrict__ x,
    const float* __restrict__ w1_dw, const float* __restrict__ b1_dw,
    const float* __restrict__ w1_pw, const float* __restrict__ b1_pw,
    _Float16* __restrict__ a1) {
    __shared__ float t[3][32][32];
    __shared__ float swp[128][4];
    const int n = blockIdx.x, tid = threadIdx.x;

    if (tid < 128) {
        swp[tid][0] = fsign(w1_pw[tid * 3 + 0]);
        swp[tid][1] = fsign(w1_pw[tid * 3 + 1]);
        swp[tid][2] = fsign(w1_pw[tid * 3 + 2]);
        swp[tid][3] = fsign(b1_pw[tid]);
    }
    const float* xn = x + (size_t)n * 3 * 1024;
    for (int i = tid; i < 3 * 1024; i += 256) {
        int c = i >> 10, p = i & 1023, y = p >> 5, xx = p & 31;
        float acc = fsign(b1_dw[c]);
        for (int ky = 0; ky < 3; ++ky) {
            int yy = y + ky - 1;
            if (yy < 0 || yy > 31) continue;
            for (int kx = 0; kx < 3; ++kx) {
                int xc = xx + kx - 1;
                if (xc < 0 || xc > 31) continue;
                acc += fsign(w1_dw[c * 9 + ky * 3 + kx]) * xn[c * 1024 + yy * 32 + xc];
            }
        }
        t[c][y][xx] = acc;
    }
    __syncthreads();
    _Float16* out = a1 + (size_t)n * 128 * 256;
    for (int i = tid; i < 32768; i += 256) {
        int oc = i >> 8, p = i & 255, y2 = p >> 4, x2 = p & 15;
        float m = -3.0e38f;
        for (int dy = 0; dy < 2; ++dy)
            for (int dx = 0; dx < 2; ++dx) {
                int y = 2 * y2 + dy, xx = 2 * x2 + dx;
                float v = swp[oc][0] * t[0][y][xx] + swp[oc][1] * t[1][y][xx] +
                          swp[oc][2] * t[2][y][xx] + swp[oc][3];
                m = fmaxf(m, v);
            }
        out[i] = (_Float16)fsign(m);   // maxpool(sign(h)) == sign(max(h))
    }
}

// ---------------------------------------------------------------------------
// K2: depthwise 3x3, g=128, pad=1 on a1.
//     Output stored PIXEL-MAJOR: d2t[n][p][ic]  (p = y*16+x, ic fastest)
//     so the k3 WMMA B-tile becomes a single contiguous 32B load per lane.
//     ic is the fastest thread index -> coalesced stores.
// ---------------------------------------------------------------------------
__global__ __launch_bounds__(256) void k2_dw2(const _Float16* __restrict__ a1,
                                              const float* __restrict__ w2_dw,
                                              const float* __restrict__ b2_dw,
                                              _Float16* __restrict__ d2t) {
    int idx = blockIdx.x * 256 + threadIdx.x;   // < 1024*256*128
    int c = idx & 127, p = (idx >> 7) & 255, n = idx >> 15;
    int y = p >> 4, xx = p & 15;
    const _Float16* in = a1 + ((size_t)n * 128 + c) * 256;
    float acc = fsign(b2_dw[c]);
    for (int ky = 0; ky < 3; ++ky) {
        int yy = y + ky - 1;
        if (yy < 0 || yy > 15) continue;
        for (int kx = 0; kx < 3; ++kx) {
            int xc = xx + kx - 1;
            if (xc < 0 || xc > 15) continue;
            acc += fsign(w2_dw[c * 9 + ky * 3 + kx]) * (float)in[yy * 16 + xc];
        }
    }
    d2t[idx] = (_Float16)acc;                   // [n][p][c]
}

// ---------------------------------------------------------------------------
// K3: conv2_pw as WMMA GEMM per image: C[oc,p] = sum_ic sign(W)[oc,ic]*d2[p,ic]
//     B element (k=ic, col=p): lane holds col p; k runs contiguously in d2t
//     row p -> one v16h (32B) load per lane per WMMA step.
//     N-columns permuted so each 16-col tile = four 2x2 pool quads; pool via
//     shfl_xor lane reduction, then sign -> a2 [N][256][8][8] (fc1 layout).
//     8 waves/block, 1 tile/wave; EXEC full (no divergence before WMMA).
// ---------------------------------------------------------------------------
__global__ __launch_bounds__(256) void k3_pw2(const _Float16* __restrict__ d2t,
                                              const _Float16* __restrict__ w2s,
                                              const float* __restrict__ b2_pw,
                                              _Float16* __restrict__ a2) {
    const int wave = threadIdx.x >> 5, lane = threadIdx.x & 31;
    const int lo = lane & 15, hi = lane >> 4;
    const int gw = blockIdx.x * 8 + wave;       // 0 .. 1024*256-1
    const int n = gw >> 8;
    const int tile = gw & 255;
    const int oc0 = (tile >> 4) * 16;
    const int t = tile & 15;
    // pooled-output index q and 2x2 subpixel for this lane's column
    const int q = t * 4 + (lo >> 2);            // q = y2*8 + x2
    const int s = lo & 3;
    const int y2 = q >> 3, x2 = q & 7;
    const int p = (2 * y2 + (s >> 1)) * 16 + 2 * x2 + (s & 1);

    const _Float16* Drow = d2t + ((size_t)n * 256 + p) * 128;  // 128 ic contiguous
    v8f acc = {};
    for (int kb = 0; kb < 128; kb += 32) {
        // A tile: rows oc0+lo of sign(w2_pw); lanes 0-15 K=0-15, lanes 16-31 K=16-31
        const _Float16* ar = w2s + (size_t)(oc0 + lo) * 128 + kb;
        v16h a = *(const v16h*)(ar + hi * 16);
        // B tile: lanes 0-15 hold K=0-15, lanes 16-31 K=16-31 -> one 32B load
        v16h b = *(const v16h*)(Drow + kb + hi * 16);
        acc = __builtin_amdgcn_wmma_f32_16x16x32_f16(false, a, false, b,
                                                     (short)0, acc, false, false);
    }
#pragma unroll
    for (int v = 0; v < 8; ++v) {
        const int oc = oc0 + v + 8 * hi;        // C/D layout: m = v + 8*hi, n = lo
        float val = acc[v] + fsign(b2_pw[oc]);
        // 2x2 max-pool across the 4 lanes sharing q (lo^1, lo^2 keep hi bit)
        val = fmaxf(val, __shfl_xor(val, 1));
        val = fmaxf(val, __shfl_xor(val, 2));
        if (s == 0)
            a2[((size_t)n * 256 + oc) * 64 + q] = (_Float16)fsign(val);
    }
}

// ---------------------------------------------------------------------------
// K6: fc1 WMMA GEMM: C[b,o] = sum_k a2[b,k] * fc1_w[o,k]; M=N=1024, K=16384
//     then bias + sign -> s1 [1024][1024] f16
// ---------------------------------------------------------------------------
__global__ __launch_bounds__(256) void k6_fc1(const _Float16* __restrict__ A2,
                                              const _Float16* __restrict__ Wh,
                                              const float* __restrict__ fc1_b,
                                              _Float16* __restrict__ s1) {
    const int wave = threadIdx.x >> 5, lane = threadIdx.x & 31;
    const int lo = lane & 15, hi = lane >> 4;
    const int gw = blockIdx.x * 8 + wave;       // 0..4095
    const int b0 = (gw >> 6) * 16;
    const int o0 = (gw & 63) * 16;
    const _Float16* arow = A2 + (size_t)(b0 + lo) * 16384;
    const _Float16* brow = Wh + (size_t)(o0 + lo) * 16384;
    v8f acc = {};
    for (int kb = 0; kb < 16384; kb += 32) {
        __builtin_prefetch(brow + kb + 512, 0, 0);   // global_prefetch_b8, 1KB ahead
        v16h a = *(const v16h*)(arow + kb + hi * 16);
        v16h b = *(const v16h*)(brow + kb + hi * 16);
        acc = __builtin_amdgcn_wmma_f32_16x16x32_f16(false, a, false, b,
                                                     (short)0, acc, false, false);
    }
    const float bias = fc1_b[o0 + lo];
#pragma unroll
    for (int v = 0; v < 8; ++v) {
        const int bb = b0 + v + 8 * hi;
        s1[(size_t)bb * 1024 + o0 + lo] = (_Float16)fsign(acc[v] + bias);
    }
}

// ---------------------------------------------------------------------------
// K7: fc2 (1024x10), fp32 VALU
// ---------------------------------------------------------------------------
__global__ __launch_bounds__(256) void k7_fc2(const _Float16* __restrict__ s1,
                                              const float* __restrict__ fc2_w,
                                              const float* __restrict__ fc2_b,
                                              float* __restrict__ out) {
    int i = blockIdx.x * 256 + threadIdx.x;     // < 10240
    if (i >= 10240) return;
    int b = i / 10, j = i - b * 10;
    const _Float16* S = s1 + (size_t)b * 1024;
    const float* W = fc2_w + (size_t)j * 1024;
    float acc = fc2_b[j];
    for (int o = 0; o < 1024; ++o) acc += (float)S[o] * W[o];
    out[i] = acc;
}

// ---------------------------------------------------------------------------
extern "C" void kernel_launch(void* const* d_in, const int* in_sizes, int n_in,
                              void* d_out, int out_size, void* d_ws, size_t ws_size,
                              hipStream_t stream) {
    const float* x     = (const float*)d_in[0];
    const float* w1_dw = (const float*)d_in[1];
    const float* b1_dw = (const float*)d_in[2];
    const float* w1_pw = (const float*)d_in[3];
    const float* b1_pw = (const float*)d_in[4];
    const float* w2_dw = (const float*)d_in[5];
    const float* b2_dw = (const float*)d_in[6];
    const float* w2_pw = (const float*)d_in[7];
    const float* b2_pw = (const float*)d_in[8];
    const float* fc1_w = (const float*)d_in[9];
    const float* fc1_b = (const float*)d_in[10];
    const float* fc2_w = (const float*)d_in[11];
    const float* fc2_b = (const float*)d_in[12];
    float* out = (float*)d_out;

    char* ws = (char*)d_ws;
    _Float16* a1  = (_Float16*)(ws);                     // 1024*128*256   f16 = 64 MiB
    _Float16* d2t = (_Float16*)(ws + 67108864ull);       // 1024*256*128   f16 = 64 MiB
    _Float16* a2  = (_Float16*)(ws + 134217728ull);      // 1024*256*64    f16 = 32 MiB
    _Float16* wh  = (_Float16*)(ws + 167772160ull);      // 1024*16384     f16 = 32 MiB
    _Float16* s1  = (_Float16*)(ws + 201326592ull);      // 1024*1024      f16 =  2 MiB
    _Float16* w2s = (_Float16*)(ws + 203423744ull);      // 256*128        f16 = 64 KiB

    k0_signw2<<<128, 256, 0, stream>>>(w2_pw, w2s);
    k5_cvt<<<65536, 256, 0, stream>>>(fc1_w, wh, 1024 * 16384);
    k1_conv1<<<1024, 256, 0, stream>>>(x, w1_dw, b1_dw, w1_pw, b1_pw, a1);
    k2_dw2<<<131072, 256, 0, stream>>>(a1, w2_dw, b2_dw, d2t);
    k3_pw2<<<32768, 256, 0, stream>>>(d2t, w2s, b2_pw, a2);
    k6_fc1<<<512, 256, 0, stream>>>(a2, wh, fc1_b, s1);
    k7_fc2<<<40, 256, 0, stream>>>(s1, fc2_w, fc2_b, out);
}
